// QuantumSeries_42803644072033
// MI455X (gfx1250) — matrix operand order repaired and checked by reading, not compiled, not run on hardware
//
#include <hip/hip_runtime.h>

#define B_DIM 8
#define L_DIM 1024
#define D_DIM 512
#define KS    25
#define PAD   12            // (KS-1)/2
#define TL    16            // L rows per workgroup
#define TROWS 40            // TL + 2*PAD
#define KPAD  64            // K padded to 2 WMMA steps of 32
#define DT_F  0.001f

typedef __attribute__((ext_vector_type(16))) _Float16      v16h;
typedef __attribute__((ext_vector_type(2)))  _Float16      v2h;
typedef __attribute__((ext_vector_type(8)))  float         v8f;
typedef __attribute__((ext_vector_type(4))) unsigned int   u32x4;
typedef __attribute__((ext_vector_type(8)))  int           i32x8;
typedef __attribute__((ext_vector_type(4)))  int           i32x4;

__global__ void __launch_bounds__(256)
QuantumSeries_kernel(const float* __restrict__ psi, float* __restrict__ out) {
    extern __shared__ char smem[];
    float*    sF32  = (float*)smem;                                  // TROWS x 512 f32 (80KB)
    _Float16* sF16T = (_Float16*)(smem + TROWS * D_DIM * 4);         // 512 x KPAD f16, transposed (64KB)
    float*    sDen  = (float*)(smem + TROWS * D_DIM * 4);            // alias: TL x 512 f32 (32KB)

    const int tid  = threadIdx.x;
    const int lane = tid & 31;
    const int wave = tid >> 5;
    const int lt   = blockIdx.x;          // 0..63  L tile
    const int b    = blockIdx.y;          // 0..7   batch
    const int l0   = lt * TL;
    const int base = l0 - PAD;

    // ---- Stage halo tile Global->LDS via Tensor Data Mover (async) ----
    int gstart = base < 0 ? 0 : base;
    int gend   = l0 + TL - 1 + PAD; if (gend > L_DIM - 1) gend = L_DIM - 1;
    int nrows  = gend - gstart + 1;
    int dstRow = gstart - base;

    if (tid == 0) {
        unsigned long long ga =
            (unsigned long long)(uintptr_t)(psi + ((size_t)b * L_DIM + gstart) * D_DIM);
        unsigned int ldsAddr =
            (unsigned int)(uintptr_t)(sF32 + (size_t)dstRow * D_DIM);
        u32x4 g0;
        g0[0] = 1u;                                            // count=1, user descriptor
        g0[1] = ldsAddr;                                       // lds_addr (bytes)
        g0[2] = (unsigned int)(ga & 0xFFFFFFFFu);              // global_addr[31:0]
        g0[3] = (unsigned int)((ga >> 32) & 0x1FFFFFFu)        // global_addr[56:32]
              | (2u << 30);                                    // type=2 ("image")
        i32x8 g1;
        g1[0] = (int)(2u << 16);                               // data_size=2 -> 4 bytes
        g1[1] = (int)((unsigned)(D_DIM & 0xFFFF) << 16);       // tensor_dim0 lo16 = 512
        g1[2] = (int)(((unsigned)nrows & 0xFFFFu) << 16);      // dim0 hi=0 | tensor_dim1 lo16
        g1[3] = (int)((unsigned)(D_DIM & 0xFFFF) << 16);       // dim1 hi=0 | tile_dim0 = 512
        g1[4] = nrows;                                         // tile_dim1 | tile_dim2=0
        g1[5] = D_DIM;                                         // tensor_dim0_stride lo32
        g1[6] = 0;                                             // stride hi | dim1_stride lo
        g1[7] = 0;
        i32x4 z4 = {0, 0, 0, 0};                               // 2D tensor: groups 2/3 unused
        i32x8 z8 = {0, 0, 0, 0, 0, 0, 0, 0};
        __builtin_amdgcn_tensor_load_to_lds(g0, g1, z4, z4, z8, 0);
    }
    __builtin_amdgcn_s_wait_tensorcnt(0);
    __syncthreads();

    // ---- Replicate-pad edge rows (only first/last L tile do any copies) ----
    if (dstRow != 0 || nrows != TROWS) {
        for (int s = 0; s < TROWS; ++s) {
            int gs = base + s;
            int cg = gs < 0 ? 0 : (gs > L_DIM - 1 ? L_DIM - 1 : gs);
            int src = cg - base;
            if (src != s)
                for (int d = tid; d < D_DIM; d += 256)
                    sF32[s * D_DIM + d] = sF32[src * D_DIM + d];
        }
        __syncthreads();
    }

    // ---- f32 -> f16, transposed [d][K]; zero K-pad; 2 halves per 32-bit store ----
    for (int i = tid; i < D_DIM * (KPAD / 2); i += 256) {
        const int d  = i >> 5;                                  // / (KPAD/2)
        const int k0 = (i & 31) * 2;
        v2h v;
        v[0] = (k0     < TROWS) ? (_Float16)sF32[(size_t)k0       * D_DIM + d] : (_Float16)0.0f;
        v[1] = (k0 + 1 < TROWS) ? (_Float16)sF32[(size_t)(k0 + 1) * D_DIM + d] : (_Float16)0.0f;
        *(v2h*)(sF16T + (size_t)d * KPAD + k0) = v;
    }
    __syncthreads();

    // ---- Band-matrix A fragments (1/25 on band m..m+24), 16-bit A 16x32 layout ----
    const int m_row = lane & 15;
    const int khalf = (lane < 16) ? 0 : 16;
    v16h aFrag[2];
    #pragma unroll
    for (int ks = 0; ks < 2; ++ks) {
        #pragma unroll
        for (int e = 0; e < 16; ++e) {
            int k = ks * 32 + khalf + e;
            float w = (k >= m_row && k <= m_row + (KS - 1)) ? (1.0f / KS) : 0.0f;
            aFrag[ks][e] = (_Float16)w;
        }
    }

    // ---- potential = A_band x tile : 4 D-tiles per wave, 2 K-steps each ----
    // B fragment: lane column d = n0+(lane&15), K = ks*32 + (lane<16?0:16) + e
    // -> 16 contiguous halves in sF16T row d: two ds_load_b128 per fragment.
    v8f cFrag[4];
    const int colB  = lane & 15;
    const int koffB = (lane < 16) ? 0 : 16;
    #pragma unroll
    for (int t = 0; t < 4; ++t) {
        const int n0 = (wave * 4 + t) * 16;
        const _Float16* brow = sF16T + (size_t)(n0 + colB) * KPAD + koffB;
        v8f c = {};
        #pragma unroll
        for (int ks = 0; ks < 2; ++ks) {
            v16h bFrag = *(const v16h*)(brow + ks * 32);
            c = __builtin_amdgcn_wmma_f32_16x16x32_f16(false, aFrag[ks], false, bFrag,
                                                       (short)0, c, false, false);
        }
        cFrag[t] = c;
    }
    __syncthreads();     // all sF16T reads done -> region reusable as density buffer

    // ---- density = psi^2 * (1 + (DT*H)^2)^1024  (10 squarings) ----
    const int mBase = (lane >> 4) * 8;
    #pragma unroll
    for (int t = 0; t < 4; ++t) {
        const int n0 = (wave * 4 + t) * 16;
        const int d  = n0 + (lane & 15);
        const int dm = (d + D_DIM - 1) & (D_DIM - 1);
        const int dp = (d + 1) & (D_DIM - 1);
        #pragma unroll
        for (int v = 0; v < 8; ++v) {
            const int m = mBase + v;
            const float* row = sF32 + (size_t)(m + PAD) * D_DIM;
            float p   = row[d];
            float kin = -0.5f * (row[dm] - 2.0f * p + row[dp]);
            float H   = kin + cFrag[t][v];
            float x   = DT_F * H;
            float tp  = 1.0f + x * x;
            #pragma unroll
            for (int q = 0; q < 10; ++q) tp *= tp;              // ^1024
            sDen[m * D_DIM + d] = p * p * tp;
        }
    }
    __syncthreads();

    // ---- softmax over D per L-row: 16 threads/row, half-wave shuffles (wave32) ----
    const int row = tid >> 4;                                   // 0..15
    const int sub = tid & 15;
    const float* dr = sDen + row * D_DIM;
    float mx = -3.402823466e38f;
    for (int d = sub; d < D_DIM; d += 16) mx = fmaxf(mx, dr[d]);
    #pragma unroll
    for (int o = 8; o > 0; o >>= 1) mx = fmaxf(mx, __shfl_xor(mx, o, 16));
    float sum = 0.0f;
    for (int d = sub; d < D_DIM; d += 16) sum += __expf(dr[d] - mx);
    #pragma unroll
    for (int o = 8; o > 0; o >>= 1) sum += __shfl_xor(sum, o, 16);
    const float inv = 1.0f / sum;
    float* orow = out + ((size_t)b * L_DIM + (l0 + row)) * D_DIM;
    for (int d = sub; d < D_DIM; d += 16) orow[d] = __expf(dr[d] - mx) * inv;
}

extern "C" void kernel_launch(void* const* d_in, const int* in_sizes, int n_in,
                              void* d_out, int out_size, void* d_ws, size_t ws_size,
                              hipStream_t stream) {
    const float* psi = (const float*)d_in[0];
    float* out = (float*)d_out;
    (void)in_sizes; (void)n_in; (void)out_size; (void)d_ws; (void)ws_size;
    dim3 grid(L_DIM / TL, B_DIM);                               // 64 x 8 workgroups
    size_t shmem = (size_t)TROWS * D_DIM * 4 + (size_t)KPAD * D_DIM * 2;  // 144 KB
    QuantumSeries_kernel<<<grid, dim3(256), shmem, stream>>>(psi, out);
}
